// Net_19808389169583
// MI455X (gfx1250) — matrix operand order repaired and checked by reading, not compiled, not run on hardware
//
#include <hip/hip_runtime.h>

// ---------------------------------------------------------------------------
// Spiking MLP forward (T=8) for MI455X / gfx1250, wave32 + WMMA f16 +
// async global->LDS staging (double-buffered).
//   sizes: B=256, d_in=3072, nhid=4096, nlab=1024
// ---------------------------------------------------------------------------

typedef __attribute__((ext_vector_type(8)))  _Float16 v8h;
typedef __attribute__((ext_vector_type(16))) _Float16 v16h;
typedef __attribute__((ext_vector_type(8)))  float    v8f;

#define TM 64      // block tile M
#define TN 128     // block tile N
#define TK 64      // K stage depth (halves)
#define LDA_P 72   // padded LDS row stride (halves): 144 B, 16B-aligned rows
#define LDB_P 72

__device__ __forceinline__ void wait_async0() {
#if __has_builtin(__builtin_amdgcn_s_wait_asynccnt)
    __builtin_amdgcn_s_wait_asynccnt(0);
#else
    asm volatile("s_wait_asynccnt 0" ::: "memory");
#endif
}

__device__ __forceinline__ void async_copy_b128_x2(unsigned lds_addr,
                                                   unsigned long long gaddr) {
    asm volatile(
        "global_load_async_to_lds_b128 %0, %1, off offset:0\n\t"
        "global_load_async_to_lds_b128 %0, %1, off offset:16"
        :: "v"(lds_addr), "v"(gaddr) : "memory");
}

__device__ __forceinline__ void async_copy_b128_x4(unsigned lds_addr,
                                                   unsigned long long gaddr) {
    asm volatile(
        "global_load_async_to_lds_b128 %0, %1, off offset:0\n\t"
        "global_load_async_to_lds_b128 %0, %1, off offset:16\n\t"
        "global_load_async_to_lds_b128 %0, %1, off offset:32\n\t"
        "global_load_async_to_lds_b128 %0, %1, off offset:48"
        :: "v"(lds_addr), "v"(gaddr) : "memory");
}

// ------------------------------ min/max ------------------------------------
__global__ void minmax_part_kernel(const float* __restrict__ x, int n,
                                   float* __restrict__ pmin,
                                   float* __restrict__ pmax) {
    __shared__ float smn[256];
    __shared__ float smx[256];
    float mn =  __builtin_inff();
    float mx = -__builtin_inff();
    for (int i = blockIdx.x * blockDim.x + threadIdx.x; i < n;
         i += gridDim.x * blockDim.x) {
        float v = x[i];
        mn = fminf(mn, v);
        mx = fmaxf(mx, v);
    }
    smn[threadIdx.x] = mn;
    smx[threadIdx.x] = mx;
    __syncthreads();
    for (int s = 128; s > 0; s >>= 1) {
        if (threadIdx.x < s) {
            smn[threadIdx.x] = fminf(smn[threadIdx.x], smn[threadIdx.x + s]);
            smx[threadIdx.x] = fmaxf(smx[threadIdx.x], smx[threadIdx.x + s]);
        }
        __syncthreads();
    }
    if (threadIdx.x == 0) {
        pmin[blockIdx.x] = smn[0];
        pmax[blockIdx.x] = smx[0];
    }
}

__global__ void minmax_final_kernel(const float* __restrict__ pmin,
                                    const float* __restrict__ pmax,
                                    int n, float* __restrict__ mm) {
    __shared__ float smn[256];
    __shared__ float smx[256];
    float mn = (threadIdx.x < n) ? pmin[threadIdx.x] :  __builtin_inff();
    float mx = (threadIdx.x < n) ? pmax[threadIdx.x] : -__builtin_inff();
    smn[threadIdx.x] = mn;
    smx[threadIdx.x] = mx;
    __syncthreads();
    for (int s = 128; s > 0; s >>= 1) {
        if (threadIdx.x < s) {
            smn[threadIdx.x] = fminf(smn[threadIdx.x], smn[threadIdx.x + s]);
            smx[threadIdx.x] = fmaxf(smx[threadIdx.x], smx[threadIdx.x + s]);
        }
        __syncthreads();
    }
    if (threadIdx.x == 0) { mm[0] = smn[0]; mm[1] = smx[0]; }
}

// --------------------------- f32 -> f16 convert ----------------------------
__global__ void cvt_f16_kernel(const float* __restrict__ src,
                               _Float16* __restrict__ dst, int n) {
    for (int i = blockIdx.x * blockDim.x + threadIdx.x; i < n;
         i += gridDim.x * blockDim.x) {
        dst[i] = (_Float16)src[i];
    }
}

// ---------------------------- binarize input -------------------------------
__device__ __forceinline__ float hash_uniform(unsigned i, unsigned t) {
    unsigned s = i * 0x9E3779B9u ^ (t * 0x85EBCA6Bu + 0x27220A95u);
    s ^= s >> 16; s *= 0x7FEB352Du;
    s ^= s >> 15; s *= 0x846CA68Bu;
    s ^= s >> 16;
    return (float)(s >> 8) * (1.0f / 16777216.0f);   // [0,1)
}

__global__ void binarize_kernel(const float* __restrict__ x,
                                const float* __restrict__ mm,
                                _Float16* __restrict__ h, int n, unsigned t) {
    int i = blockIdx.x * blockDim.x + threadIdx.x;
    if (i >= n) return;
    float mn = mm[0], mx = mm[1];
    float r = mn + hash_uniform((unsigned)i, t) * (mx - mn);
    h[i] = (x[i] > r) ? (_Float16)1.0f : (_Float16)0.0f;
}

// --------------------------------- LIF -------------------------------------
__global__ void lif_kernel(const float* __restrict__ z,
                           const float* __restrict__ bias,
                           float* __restrict__ mem,
                           _Float16* __restrict__ spike,
                           float* __restrict__ out_acc,
                           int total, int n_mask) {
    int i = blockIdx.x * blockDim.x + threadIdx.x;
    if (i >= total) return;
    float zz = z[i];
    if (bias) zz += bias[i & n_mask];
    float s_old = (float)spike[i];
    float m = mem[i] * 0.2f * (1.0f - s_old) + 1.0f / (1.0f + __expf(-zz));
    float s = (m > 0.5f) ? 1.0f : 0.0f;
    mem[i] = m;
    spike[i] = (_Float16)s;
    if (out_acc) out_acc[i] += s * 0.125f;   // 1/T
}

// --------------------------- WMMA GEMM (C = A * W^T) -----------------------
// A: [M,K] f16 row-major   W: [N,K] f16 row-major   C: [M,N] f32 row-major
// block: 256 threads = 8 waves; each wave -> 32x32 output (2x2 WMMA tiles).
// Double-buffered LDS stages filled with GLOBAL_LOAD_ASYNC_TO_LDS_B128.
template <int N, int K>
__global__ __launch_bounds__(256)
void wmma_gemm_nt_kernel(const _Float16* __restrict__ A,
                         const _Float16* __restrict__ W,
                         float* __restrict__ C, int M) {
    __shared__ _Float16 lA[2][TM * LDA_P];   // 2 x  9216 B
    __shared__ _Float16 lB[2][TN * LDB_P];   // 2 x 18432 B

    const int tid  = threadIdx.x;
    const int lane = tid & 31;
    const int wid  = tid >> 5;       // 0..7
    const int wm   = wid >> 2;       // 0..1  (M direction)
    const int wn   = wid & 3;        // 0..3  (N direction)
    const int sel  = lane >> 4;      // lane half (K split per CDNA5 layouts)
    const int l16  = lane & 15;

    const int mBlk = blockIdx.y * TM;
    const int nBlk = blockIdx.x * TN;

    v8f acc[2][2];
#pragma unroll
    for (int i = 0; i < 2; ++i)
#pragma unroll
        for (int j = 0; j < 2; ++j) acc[i][j] = (v8f)0.0f;

    // staging assignments (per thread: A 32 B, W 64 B, contiguous)
    const int aRow = tid >> 2;         // 0..63
    const int aCol = (tid & 3) * 16;   // 0/16/32/48
    const int bRow = tid >> 1;         // 0..127
    const int bCol = (tid & 1) * 32;   // 0/32

    const _Float16* gA0 = A + (size_t)(mBlk + aRow) * K + aCol;
    const _Float16* gW0 = W + (size_t)(nBlk + bRow) * K + bCol;

    auto issue_stage = [&](int buf, int k0) {
        unsigned la = (unsigned)(size_t)(&lA[buf][aRow * LDA_P + aCol]);
        async_copy_b128_x2(la, (unsigned long long)(gA0 + k0));
        unsigned lb = (unsigned)(size_t)(&lB[buf][bRow * LDB_P + bCol]);
        async_copy_b128_x4(lb, (unsigned long long)(gW0 + k0));
    };

    auto compute_stage = [&](int buf) {
#pragma unroll
        for (int kk = 0; kk < TK; kk += 32) {
            v16h a[2], b[2];
            // A frag: lanes 0-15 hold K {0..7,16..23}, lanes 16-31 K {8..15,24..31}
#pragma unroll
            for (int i = 0; i < 2; ++i) {
                const _Float16* p =
                    &lA[buf][(wm * 32 + i * 16 + l16) * LDA_P + kk + sel * 8];
                v8h lo = *(const v8h*)p;
                v8h hi = *(const v8h*)(p + 16);
                a[i] = __builtin_shufflevector(lo, hi, 0, 1, 2, 3, 4, 5, 6, 7,
                                               8, 9, 10, 11, 12, 13, 14, 15);
            }
            // B frag: lanes 0-15 hold K {0..7,16..23}, lanes 16-31 K {8..15,24..31}
#pragma unroll
            for (int j = 0; j < 2; ++j) {
                const _Float16* p =
                    &lB[buf][(wn * 32 + j * 16 + l16) * LDB_P + kk + sel * 8];
                v8h b0 = *(const v8h*)p;
                v8h b1 = *(const v8h*)(p + 16);
                b[j] = __builtin_shufflevector(b0, b1, 0, 1, 2, 3, 4, 5, 6, 7,
                                               8, 9, 10, 11, 12, 13, 14, 15);
            }
#pragma unroll
            for (int i = 0; i < 2; ++i)
#pragma unroll
                for (int j = 0; j < 2; ++j)
                    acc[i][j] = __builtin_amdgcn_wmma_f32_16x16x32_f16(
                        false, a[i], false, b[j], (short)0, acc[i][j], false,
                        false);
        }
    };

    constexpr int NST = K / TK;      // 48 or 64: always even
    issue_stage(0, 0);
#pragma unroll 1
    for (int s = 0; s < NST; s += 2) {
        // even stage: buffer 0
        wait_async0();
        __syncthreads();                         // lA/lB[0] complete for all
        if (s + 1 < NST) issue_stage(1, (s + 1) * TK);
        compute_stage(0);
        __syncthreads();                         // all done reading buf 0
        // odd stage: buffer 1
        wait_async0();
        __syncthreads();                         // lA/lB[1] complete for all
        if (s + 2 < NST) issue_stage(0, (s + 2) * TK);
        compute_stage(1);
        __syncthreads();                         // all done reading buf 1
    }

    // C layout: VGPR r -> M = r + sel*8, N = l16 (constant N -> imm offsets)
#pragma unroll
    for (int i = 0; i < 2; ++i)
#pragma unroll
        for (int j = 0; j < 2; ++j) {
            const int gm = mBlk + wm * 32 + i * 16 + sel * 8;
            const int gn = nBlk + wn * 32 + j * 16 + l16;
            float* pc = C + (size_t)gm * N + gn;
#pragma unroll
            for (int r = 0; r < 8; ++r) pc[(size_t)r * N] = acc[i][j][r];
        }
}

// ------------------------------- launch ------------------------------------
extern "C" void kernel_launch(void* const* d_in, const int* in_sizes, int n_in,
                              void* d_out, int out_size, void* d_ws,
                              size_t ws_size, hipStream_t stream) {
    (void)in_sizes; (void)n_in; (void)ws_size;
    const int B = 256, DIN = 3072, NH = 4096, NL = 1024, T = 8;

    const float* x  = (const float*)d_in[0];
    const float* w0 = (const float*)d_in[1];
    const float* w1 = (const float*)d_in[2];
    const float* w2 = (const float*)d_in[3];
    const float* wl = (const float*)d_in[4];
    const float* bl = (const float*)d_in[5];
    float* out = (float*)d_out;

    char* ws = (char*)d_ws;
    size_t off = 0;
    auto alloc = [&](size_t bytes) {
        char* p = ws + off;
        off += (bytes + 255) & ~(size_t)255;
        return p;
    };

    _Float16* WH0 = (_Float16*)alloc((size_t)NH * DIN * 2);
    _Float16* WH1 = (_Float16*)alloc((size_t)NH * NH * 2);
    _Float16* WH2 = (_Float16*)alloc((size_t)NH * NH * 2);
    _Float16* WHL = (_Float16*)alloc((size_t)NL * NH * 2);
    _Float16* H16 = (_Float16*)alloc((size_t)B * DIN * 2);
    // zero-initialized state region (contiguous)
    char* zero_base = ws + off;
    _Float16* S0 = (_Float16*)alloc((size_t)B * NH * 2);
    _Float16* S1 = (_Float16*)alloc((size_t)B * NH * 2);
    _Float16* S2 = (_Float16*)alloc((size_t)B * NH * 2);
    _Float16* S3 = (_Float16*)alloc((size_t)B * NL * 2);
    float* M0 = (float*)alloc((size_t)B * NH * 4);
    float* M1 = (float*)alloc((size_t)B * NH * 4);
    float* M2 = (float*)alloc((size_t)B * NH * 4);
    float* M3 = (float*)alloc((size_t)B * NL * 4);
    size_t zero_bytes = (size_t)((ws + off) - zero_base);
    float* Z    = (float*)alloc((size_t)B * NH * 4);
    float* PMIN = (float*)alloc(256 * 4);
    float* PMAX = (float*)alloc(256 * 4);
    float* MM   = (float*)alloc(2 * 4);

    // reset state + output accumulator (graph-capture-safe memsets)
    hipMemsetAsync(zero_base, 0, zero_bytes, stream);
    hipMemsetAsync(out, 0, (size_t)out_size * sizeof(float), stream);

    // input min/max
    minmax_part_kernel<<<256, 256, 0, stream>>>(x, B * DIN, PMIN, PMAX);
    minmax_final_kernel<<<1, 256, 0, stream>>>(PMIN, PMAX, 256, MM);

    // weights -> f16 (halves HBM traffic; f16 exact for {0,1} activations)
    cvt_f16_kernel<<<2048, 256, 0, stream>>>(w0, WH0, NH * DIN);
    cvt_f16_kernel<<<2048, 256, 0, stream>>>(w1, WH1, NH * NH);
    cvt_f16_kernel<<<2048, 256, 0, stream>>>(w2, WH2, NH * NH);
    cvt_f16_kernel<<<2048, 256, 0, stream>>>(wl, WHL, NL * NH);

    const dim3 blk(256);
    const dim3 gHid(NH / TN, B / TM);   // (32, 4)
    const dim3 gLab(NL / TN, B / TM);   // (8, 4)

    for (int t = 0; t < T; ++t) {
        binarize_kernel<<<(B * DIN + 255) / 256, blk, 0, stream>>>(
            x, MM, H16, B * DIN, (unsigned)t);

        wmma_gemm_nt_kernel<4096, 3072><<<gHid, blk, 0, stream>>>(H16, WH0, Z, B);
        lif_kernel<<<(B * NH + 255) / 256, blk, 0, stream>>>(
            Z, nullptr, M0, S0, nullptr, B * NH, NH - 1);

        wmma_gemm_nt_kernel<4096, 4096><<<gHid, blk, 0, stream>>>(S0, WH1, Z, B);
        lif_kernel<<<(B * NH + 255) / 256, blk, 0, stream>>>(
            Z, nullptr, M1, S1, nullptr, B * NH, NH - 1);

        wmma_gemm_nt_kernel<4096, 4096><<<gHid, blk, 0, stream>>>(S1, WH2, Z, B);
        lif_kernel<<<(B * NH + 255) / 256, blk, 0, stream>>>(
            Z, nullptr, M2, S2, nullptr, B * NH, NH - 1);

        wmma_gemm_nt_kernel<1024, 4096><<<gLab, blk, 0, stream>>>(S2, WHL, Z, B);
        lif_kernel<<<(B * NL + 255) / 256, blk, 0, stream>>>(
            Z, bl, M3, S3, out, B * NL, NL - 1);
    }
}
